// FtopkLoss_27848567947598
// MI455X (gfx1250) — matrix operand order repaired and checked by reading, not compiled, not run on hardware
//
#include <hip/hip_runtime.h>

// DINO-style topk loss, restructured:
//   loss_b = sum_k p[b,k] * ( LSE(P[idx]/0.1) - 10*dot(f_b, P[idx]) )
// Dots done with v_wmma_f32_16x16x32_f16 (A = 8 gathered predictor rows,
// B = 7 student softmax rows recomputed on the fly from row stats).
// Memory-bound problem; all gathered rows / student logits are L2-resident.

typedef _Float16 v16h __attribute__((ext_vector_type(16)));
typedef float    v8f  __attribute__((ext_vector_type(8)));

#define D_DIM  8192
#define B_SZ   256
#define NCROPS 8
#define TOPK   8
#define THREADS 256
#define ELEMS  (D_DIM / THREADS)   // 32

// ---------- Kernel A: per student row, max and sum-exp of (x*10) ----------
__global__ __launch_bounds__(THREADS) void k_student_stats(
    const float* __restrict__ S, float* __restrict__ s_max, float* __restrict__ s_z)
{
  __shared__ float red[THREADS];
  const int row = blockIdx.x;
  const int tid = threadIdx.x;
  const float* x = S + (size_t)row * D_DIM;
  float v[ELEMS];
  float m = -INFINITY;
#pragma unroll
  for (int j = 0; j < ELEMS; ++j) { v[j] = x[tid + j*THREADS] * 10.0f; m = fmaxf(m, v[j]); }
  red[tid] = m; __syncthreads();
  for (int s = THREADS/2; s > 0; s >>= 1) { if (tid < s) red[tid] = fmaxf(red[tid], red[tid+s]); __syncthreads(); }
  m = red[0]; __syncthreads();
  float z = 0.f;
#pragma unroll
  for (int j = 0; j < ELEMS; ++j) z += expf(v[j] - m);
  red[tid] = z; __syncthreads();
  for (int s = THREADS/2; s > 0; s >>= 1) { if (tid < s) red[tid] += red[tid+s]; __syncthreads(); }
  if (tid == 0) { s_max[row] = m; s_z[row] = red[0]; }
}

// ---------- Kernel B: teacher top-8 (indices + normalized p) per row ----------
__global__ __launch_bounds__(THREADS) void k_teacher_topk(
    const float* __restrict__ T, const float* __restrict__ center,
    const int* __restrict__ epoch_p,
    float* __restrict__ t_p, int* __restrict__ t_idx)
{
  __shared__ float sval[THREADS];
  __shared__ int   sidx[THREADS];
  __shared__ float sel_val[TOPK];
  __shared__ int   sel_idx[TOPK];
  const int row = blockIdx.x;
  const int tid = threadIdx.x;
  const int ep = epoch_p[0];
  const float temp = (ep < 30) ? (0.04f + 0.03f * (float)ep / 29.0f) : 0.07f;
  const float invt = 1.0f / temp;
  const float* x = T + (size_t)row * D_DIM;
  float z[ELEMS];
#pragma unroll
  for (int j = 0; j < ELEMS; ++j) {
    const int d = tid + j*THREADS;
    z[j] = (x[d] - center[d]) * invt;
  }
  unsigned used = 0u;
  for (int k = 0; k < TOPK; ++k) {
    float bv = -INFINITY; int bi = 0x7fffffff;
#pragma unroll
    for (int j = 0; j < ELEMS; ++j) {
      if (used & (1u << j)) continue;
      const int d = tid + j*THREADS;
      if (z[j] > bv || (z[j] == bv && d < bi)) { bv = z[j]; bi = d; }
    }
    sval[tid] = bv; sidx[tid] = bi; __syncthreads();
    for (int s = THREADS/2; s > 0; s >>= 1) {
      if (tid < s) {
        const float ov = sval[tid+s]; const int oi = sidx[tid+s];
        if (ov > sval[tid] || (ov == sval[tid] && oi < sidx[tid])) { sval[tid] = ov; sidx[tid] = oi; }
      }
      __syncthreads();
    }
    const int w = sidx[0];
    if (tid == 0) { sel_val[k] = sval[0]; sel_idx[k] = w; }
    if ((w & (THREADS-1)) == tid) used |= (1u << (w >> 8));
    __syncthreads();
  }
  if (tid == 0) {
    const float v0 = sel_val[0];
    float pv[TOPK]; float sum = 0.f;
    for (int k = 0; k < TOPK; ++k) { pv[k] = expf(sel_val[k] - v0); sum += pv[k]; }
    const float inv = 1.0f / sum;
    for (int k = 0; k < TOPK; ++k) {
      t_p[row*TOPK + k]   = pv[k] * inv;
      t_idx[row*TOPK + k] = sel_idx[k];
    }
  }
}

// ---------- Kernel C1: LSE(P[r,:]*10) for each gathered row ----------
__global__ __launch_bounds__(THREADS) void k_lse(
    const float* __restrict__ P, const int* __restrict__ t_idx, float* __restrict__ lse)
{
  __shared__ float red[THREADS];
  const int e = blockIdx.x;            // 0..4095
  const int tid = threadIdx.x;
  const int r = t_idx[e];
  const float* x = P + (size_t)r * D_DIM;
  float v[ELEMS]; float m = -INFINITY;
#pragma unroll
  for (int j = 0; j < ELEMS; ++j) { v[j] = x[tid + j*THREADS] * 10.0f; m = fmaxf(m, v[j]); }
  red[tid] = m; __syncthreads();
  for (int s = THREADS/2; s > 0; s >>= 1) { if (tid < s) red[tid] = fmaxf(red[tid], red[tid+s]); __syncthreads(); }
  m = red[0]; __syncthreads();
  float zz = 0.f;
#pragma unroll
  for (int j = 0; j < ELEMS; ++j) zz += expf(v[j] - m);
  red[tid] = zz; __syncthreads();
  for (int s = THREADS/2; s > 0; s >>= 1) { if (tid < s) red[tid] += red[tid+s]; __syncthreads(); }
  if (tid == 0) lse[e] = logf(red[0]) + m;
}

// ---------- Kernel D: column mean of teacher_output ----------
__global__ __launch_bounds__(THREADS) void k_batch_center(
    const float* __restrict__ T, float* __restrict__ bc)
{
  const int col = blockIdx.x * THREADS + threadIdx.x;
  float s = 0.f;
  for (int r = 0; r < 2*B_SZ; ++r) s += T[(size_t)r * D_DIM + col];
  bc[col] = s * (1.0f / (2*B_SZ));
}

// ---------- Kernel C2: WMMA dot-matrix per (iq,b) + loss partials ----------
// A (16x32 f16): rows 0..7 = gathered predictor rows, rows 8..15 zero.
// B (32x16 f16): cols 1..7 = student softmax rows f_{v,b} (recomputed).
// 8 waves partition K=8192; LDS reduction of C fragments.
__global__ __launch_bounds__(256) void k_dots(
    const float* __restrict__ P, const float* __restrict__ S,
    const int* __restrict__ t_idx, const float* __restrict__ t_p,
    const float* __restrict__ lse,
    const float* __restrict__ s_max, const float* __restrict__ s_z,
    float* __restrict__ loss_part)
{
  __shared__ float red[8][32][8];
  const int blk  = blockIdx.x;        // 0..511 == teacher row (iq*256+b)
  const int iq   = blk >> 8;
  const int b    = blk & 255;
  const int tid  = threadIdx.x;
  const int wave = tid >> 5;
  const int lane = tid & 31;
  const int m    = lane & 15;
  const int hi   = lane >> 4;
  const int base = blk * TOPK;

  const bool mvalid = (m < TOPK);
  const int r = mvalid ? t_idx[base + m] : 0;
  const float* prow = P + (size_t)r * D_DIM;

  const int n = m;                     // student crop index (column)
  const bool nvalid = (n >= 1 && n < NCROPS);
  const float* srow = S + (size_t)(n * B_SZ + b) * D_DIM;
  float smax = 0.f, sinv = 0.f;
  if (nvalid) { smax = s_max[n*B_SZ + b]; sinv = 1.0f / s_z[n*B_SZ + b]; }

  const int akb = hi ? 8 : 0;          // A: lane<16 K in {0..7,16..23}, else {8..15,24..31}
  const int bkb = hi ? 8 : 0;          // B: lane<16 K in {0..7,16..23}, else {8..15,24..31}

  v8f acc = {0.f,0.f,0.f,0.f,0.f,0.f,0.f,0.f};
  for (int i = 0; i < 32; ++i) {
    const int k0 = (i * 8 + wave) * 32;
    v16h af, bf;
    if (mvalid) {
      const float4* p0 = (const float4*)(prow + k0 + akb);
      const float4 x0 = p0[0], x1 = p0[1], x2 = p0[4], x3 = p0[5];
      float av[16];
      av[0]=x0.x; av[1]=x0.y; av[2]=x0.z; av[3]=x0.w;
      av[4]=x1.x; av[5]=x1.y; av[6]=x1.z; av[7]=x1.w;
      av[8]=x2.x; av[9]=x2.y; av[10]=x2.z; av[11]=x2.w;
      av[12]=x3.x; av[13]=x3.y; av[14]=x3.z; av[15]=x3.w;
#pragma unroll
      for (int e2 = 0; e2 < 16; ++e2) af[e2] = (_Float16)av[e2];
    } else {
#pragma unroll
      for (int e2 = 0; e2 < 16; ++e2) af[e2] = (_Float16)0.0f;
    }
    if (nvalid) {
      const float4* q0 = (const float4*)(srow + k0 + bkb);
      const float4 y0 = q0[0], y1 = q0[1], y2 = q0[4], y3 = q0[5];
      float fv[16];
      fv[0]=y0.x; fv[1]=y0.y; fv[2]=y0.z; fv[3]=y0.w;
      fv[4]=y1.x; fv[5]=y1.y; fv[6]=y1.z; fv[7]=y1.w;
      fv[8]=y2.x; fv[9]=y2.y; fv[10]=y2.z; fv[11]=y2.w;
      fv[12]=y3.x; fv[13]=y3.y; fv[14]=y3.z; fv[15]=y3.w;
#pragma unroll
      for (int e2 = 0; e2 < 16; ++e2)
        bf[e2] = (_Float16)(expf(fv[e2]*10.0f - smax) * sinv);
    } else {
#pragma unroll
      for (int e2 = 0; e2 < 16; ++e2) bf[e2] = (_Float16)0.0f;
    }
    acc = __builtin_amdgcn_wmma_f32_16x16x32_f16(
        false, af, false, bf, (short)0, acc, false, false);
  }
#pragma unroll
  for (int c = 0; c < 8; ++c) red[wave][lane][c] = acc[c];
  __syncthreads();
  if (wave == 0 && lane < 16) {
    float c8[8];
#pragma unroll
    for (int c = 0; c < 8; ++c) {
      float s2 = 0.f;
      for (int w = 0; w < 8; ++w) s2 += red[w][lane][c];
      c8[c] = s2;                      // c8[k] = dot(P[idx_k], f_{v=lane,b})
    }
    float part = 0.f;
    const int vlo = (iq == 0) ? 1 : 2;
    if (lane >= vlo && lane < NCROPS) {
      for (int k = 0; k < TOPK; ++k)
        part += t_p[base + k] * (lse[base + k] - 10.0f * c8[k]);
    }
    loss_part[blk * 16 + lane] = part;
  }
}

// ---------- Kernel E: final deterministic reductions + center/entropies ----------
__global__ __launch_bounds__(THREADS) void k_final(
    const float* __restrict__ loss_part,
    const float* __restrict__ center,
    const float* __restrict__ bc,
    float* __restrict__ out)
{
  __shared__ float red[THREADS];
  const int tid = threadIdx.x;
  // total_loss
  float s = 0.f;
  for (int i = 0; i < 32; ++i) s += loss_part[tid + i*THREADS];
  red[tid] = s; __syncthreads();
  for (int st = THREADS/2; st > 0; st >>= 1) { if (tid < st) red[tid] += red[tid+st]; __syncthreads(); }
  if (tid == 0) out[0] = red[0] / (13.0f * 256.0f);
  __syncthreads();

  float c[ELEMS], q[ELEMS];
#pragma unroll
  for (int j = 0; j < ELEMS; ++j) { const int d = tid + j*THREADS; c[j] = center[d]; q[j] = bc[d]; }

  // LSE(center)
  float m = -INFINITY;
#pragma unroll
  for (int j = 0; j < ELEMS; ++j) m = fmaxf(m, c[j]);
  red[tid] = m; __syncthreads();
  for (int st = THREADS/2; st > 0; st >>= 1) { if (tid < st) red[tid] = fmaxf(red[tid], red[tid+st]); __syncthreads(); }
  const float mc = red[0]; __syncthreads();
  float z = 0.f;
#pragma unroll
  for (int j = 0; j < ELEMS; ++j) z += expf(c[j] - mc);
  red[tid] = z; __syncthreads();
  for (int st = THREADS/2; st > 0; st >>= 1) { if (tid < st) red[tid] += red[tid+st]; __syncthreads(); }
  const float lse_c = logf(red[0]) + mc; __syncthreads();

  // LSE(batch_center)
  m = -INFINITY;
#pragma unroll
  for (int j = 0; j < ELEMS; ++j) m = fmaxf(m, q[j]);
  red[tid] = m; __syncthreads();
  for (int st = THREADS/2; st > 0; st >>= 1) { if (tid < st) red[tid] = fmaxf(red[tid], red[tid+st]); __syncthreads(); }
  const float mbc = red[0]; __syncthreads();
  z = 0.f;
#pragma unroll
  for (int j = 0; j < ELEMS; ++j) z += expf(q[j] - mbc);
  red[tid] = z; __syncthreads();
  for (int st = THREADS/2; st > 0; st >>= 1) { if (tid < st) red[tid] += red[tid+st]; __syncthreads(); }
  const float lse_bc = logf(red[0]) + mbc; __syncthreads();

  // true_entropy = sum softmax(center)*logsm(center); entropy uses softmax(bc)
  float te = 0.f, en = 0.f;
#pragma unroll
  for (int j = 0; j < ELEMS; ++j) {
    const float lc = c[j] - lse_c;
    te += expf(lc) * lc;
    en += expf(q[j] - lse_bc) * lc;
  }
  red[tid] = te; __syncthreads();
  for (int st = THREADS/2; st > 0; st >>= 1) { if (tid < st) red[tid] += red[tid+st]; __syncthreads(); }
  if (tid == 0) out[2] = red[0];
  __syncthreads();
  red[tid] = en; __syncthreads();
  for (int st = THREADS/2; st > 0; st >>= 1) { if (tid < st) red[tid] += red[tid+st]; __syncthreads(); }
  if (tid == 0) out[1] = red[0];

  // new_center
#pragma unroll
  for (int j = 0; j < ELEMS; ++j) {
    const int d = tid + j*THREADS;
    out[3 + d] = 0.9f * c[j] + 0.1f * q[j];
  }
}

extern "C" void kernel_launch(void* const* d_in, const int* in_sizes, int n_in,
                              void* d_out, int out_size, void* d_ws, size_t ws_size,
                              hipStream_t stream)
{
  (void)in_sizes; (void)n_in; (void)out_size; (void)ws_size;
  const float* S = (const float*)d_in[0];   // student_output [2048, 8192]
  const float* T = (const float*)d_in[1];   // teacher_output [512, 8192]
  const float* P = (const float*)d_in[2];   // predictor_table [8192, 8192]
  const float* C = (const float*)d_in[3];   // center [8192]
  const int*   E = (const int*)d_in[4];     // epoch

  float* W     = (float*)d_ws;              // 32768 floats = 128 KB total
  float* s_max = W;                         // 2048
  float* s_z   = W + 2048;                  // 2048
  float* bc    = W + 4096;                  // 8192
  float* t_p   = W + 12288;                 // 4096
  int*   t_idx = (int*)(W + 16384);         // 4096
  float* lse   = W + 20480;                 // 4096
  float* lossp = W + 24576;                 // 8192
  float* out   = (float*)d_out;

  k_student_stats<<<2048, 256, 0, stream>>>(S, s_max, s_z);
  k_teacher_topk <<<512,  256, 0, stream>>>(T, C, E, t_p, t_idx);
  k_batch_center <<<32,   256, 0, stream>>>(T, bc);
  k_lse          <<<4096, 256, 0, stream>>>(P, t_idx, lse);
  k_dots         <<<512,  256, 0, stream>>>(P, S, t_idx, t_p, lse, s_max, s_z, lossp);
  k_final        <<<1,    256, 0, stream>>>(lossp, C, bc, out);
}
